// PixelCNN_8693013807800
// MI455X (gfx1250) — matrix operand order repaired and checked by reading, not compiled, not run on hardware
//
#include <hip/hip_runtime.h>

// ---------------------------------------------------------------------------
// Fused PixelCNN forward for MI455X (gfx1250, wave32, WMMA).
//   conv7x7 masked (1->16) -> relu -> conv3x3 masked (16->16) -> relu
//   -> conv1x1 (16->1) -> sigmoid
// One block per image. x staged in LDS as sliding-window half2 pairs so every
// conv1 im2col K-pair is one aligned ds_load_b32 (merged to ds_load_2addr);
// h1 staged [pixel][ch] so every conv2 tap is one ds_load_b128. Biases ride
// in on the WMMA C operand; ReLU is v_med3_f32; conv3 channel-dot is a DPP
// butterfly (no LDS); output written as float4 global_store_b128.
// ---------------------------------------------------------------------------

typedef __attribute__((ext_vector_type(16))) _Float16     v16h;
typedef __attribute__((ext_vector_type(8)))  float        v8f;
typedef __attribute__((ext_vector_type(8)))  unsigned int v8u;

#define XST2      72                      // row stride of pair tile (dwords)
#define XROWS     70                      // 64 + 2*3 halo
#define IMG       64
#define XS2_BYTES (XROWS * XST2 * 4)      // 20160
#define H1_BYTES  (IMG * IMG * 16 * 2)    // 131072

// dpp8 lane-selector constants (within each 8-lane group)
#define SEL_XOR1 ((1<<0)|(0<<3)|(3<<6)|(2<<9)|(5<<12)|(4<<15)|(7<<18)|(6<<21))
#define SEL_XOR2 ((2<<0)|(3<<3)|(0<<6)|(1<<9)|(6<<12)|(7<<15)|(4<<18)|(5<<21))
#define SEL_XOR4 ((4<<0)|(5<<3)|(6<<6)|(7<<9)|(0<<12)|(1<<15)|(2<<18)|(3<<21))

union HU { v16h h; v8u u; };

__device__ __forceinline__ unsigned int pkh(_Float16 a, _Float16 b) {
    union { _Float16 h[2]; unsigned int u; } z;
    z.h[0] = a; z.h[1] = b; return z.u;
}

__device__ __forceinline__ float relu1(float x) {          // single v_med3_f32
    return __builtin_amdgcn_fmed3f(x, 0.0f, 3.0e38f);
}

template<int SEL>
__device__ __forceinline__ float dpp8f(float v) {          // value from peer lane
    union { float f; int i; } a, r;
    a.f = v;
    r.i = __builtin_amdgcn_mov_dpp8(a.i, SEL);
    return r.f;
}

__device__ __forceinline__ float ror8_addf(float v) {      // += lane (i+8)%16
    union { float f; int i; } a, r;
    a.f = v;
    r.i = __builtin_amdgcn_update_dpp(0, a.i, 0x128 /*row_ror:8*/, 0xF, 0xF, true);
    return v + r.f;
}

__device__ __forceinline__ uint4 ldt(const char* p, int off, bool pred) {
    uint4 q = make_uint4(0u, 0u, 0u, 0u);
    if (pred) q = *(const uint4*)(p + off);   // ds_load_b128 under exec mask
    return q;
}

__global__ void pixelcnn_fused(const float* __restrict__ x,
                               const float* __restrict__ w1,
                               const float* __restrict__ b1,
                               const float* __restrict__ w2,
                               const float* __restrict__ b2,
                               const float* __restrict__ w3,
                               const float* __restrict__ b3,
                               float* __restrict__ out)
{
    extern __shared__ __align__(16) char smem[];
    unsigned int* xs2 = (unsigned int*)smem;             // [70][72] half2 pairs
    _Float16*     h1  = (_Float16*)(smem + XS2_BYTES);   // [4096][16] f16 acts

    const int tid  = threadIdx.x;
    const int lane = tid & 31;
    const int wv   = tid >> 5;        // wave id 0..7 -> rows [8*wv, 8*wv+8)
    const int b    = blockIdx.x;      // image index
    const int n    = lane & 15;       // N (out channel) for B/C/D fragments
    const int grp  = lane >> 4;       // half-wave group
    const int m    = n;               // A-matrix row (pixel within tile)

    // ---- conv1 B fragment. K -> pair p = K>>1, elem e = K&1;
    //      pair p -> tap (ty = p>>2, tx = (p&3)*2 + e); dead taps get 0 weight.
    v16h bw1;
    #pragma unroll
    for (int v = 0; v < 8; ++v) {
        #pragma unroll
        for (int e = 0; e < 2; ++e) {
            int p  = grp * 8 + v;
            int ty = p >> 2;
            int tx = ((p & 3) << 1) + e;
            float wval = 0.0f;
            if ((ty < 3 && tx < 7) || (ty == 3 && tx < 3))   // 24 live causal taps
                wval = w1[n * 49 + ty * 7 + tx];
            bw1[v * 2 + e] = (_Float16)wval;
        }
    }
    // ---- conv2 B fragments. K = tap*16 + ch; taps 0..4 live, tap 5 zero pad.
    v16h bw2[3];
    #pragma unroll
    for (int j = 0; j < 3; ++j) {
        #pragma unroll
        for (int v = 0; v < 8; ++v) {
            #pragma unroll
            for (int e = 0; e < 2; ++e) {
                int kg  = j * 32 + ((v & 4) << 2) + grp * 8 + (v & 3) * 2 + e;
                int tap = kg >> 4, ch = kg & 15;
                float wval = 0.0f;
                if (tap < 5) {
                    int ty = (tap < 3) ? 0 : 1;
                    int tx = (tap < 3) ? tap : (tap - 3);
                    wval = w2[n * 144 + ch * 9 + ty * 3 + tx];
                }
                bw2[j][v * 2 + e] = (_Float16)wval;
            }
        }
    }
    const float b1v = b1[n], b2v = b2[n], w3v = w3[n], b3s = b3[0];
    v8f cb1, cb2;                      // bias broadcast as WMMA C operand
    #pragma unroll
    for (int v = 0; v < 8; ++v) { cb1[v] = b1v; cb2[v] = b2v; }

    // ---- per-lane conv1 gather offsets (hoisted).
    int off1[8];
    #pragma unroll
    for (int v = 0; v < 8; ++v) {
        int p = (grp << 3) + v;
        off1[v] = ((p >> 2) * XST2 + ((p & 3) << 1)) * 4;
    }

    // ---- stage 0: x (fp32) -> LDS sliding-window half2 pairs, zero halo ---
    for (int i = tid; i < XROWS * XST2; i += 256) xs2[i] = 0u;
    __syncthreads();
    const float4* xg4 = (const float4*)(x + (size_t)b * (IMG * IMG));
    for (int i = tid; i < (IMG * IMG) / 4; i += 256) {
        float4 q = xg4[i];
        int r = i >> 4, c = (i & 15) << 2;
        _Float16 e0 = (_Float16)q.x, e1 = (_Float16)q.y,
                 e2 = (_Float16)q.z, e3 = (_Float16)q.w;
        unsigned int* row = xs2 + (r + 3) * XST2 + (c + 3);
        row[0] = pkh(e0, e1);
        row[1] = pkh(e1, e2);
        row[2] = pkh(e2, e3);
        _Float16* rh = (_Float16*)xs2;
        rh[((r + 3) * XST2 + (c + 6)) * 2]     = e3;   // lo half of pair c+6
        rh[((r + 3) * XST2 + (c + 2)) * 2 + 1] = e0;   // hi half of pair c+2
    }
    __syncthreads();

    // ---- stage 1: conv1 7x7 (1->16), one WMMA per 16-pixel tile ----------
    const char* xb = (const char*)xs2;
    #pragma unroll 2
    for (int t = 0; t < 32; ++t) {
        int r  = wv * 8 + (t >> 2);
        int c0 = (t & 3) << 4;
        int c  = c0 + m;
        const char* basex = xb + (r * XST2 + c) * 4;   // haloed window top-left
        HU a;
        #pragma unroll
        for (int v = 0; v < 8; ++v)
            a.u[v] = *(const unsigned int*)(basex + off1[v]);
        v8f acc = __builtin_amdgcn_wmma_f32_16x16x32_f16(false, a.h, false, bw1,
                                                         (short)0, cb1, false, false);
        // relu (v_med3) -> f16 -> transposed b16 stores (imm-offset merged)
        #pragma unroll
        for (int v = 0; v < 8; ++v) {
            float f = relu1(acc[v]);
            h1[((r * IMG) + c0 + v + grp * 8) * 16 + n] = (_Float16)f;
        }
    }
    __syncthreads();

    // ---- stage 2: conv2 3x3 (16->16) = 3 chained WMMAs, 1 b128 per tap ---
    // Live taps (dy,dx): (-1,-1)(-1,0)(-1,1)(0,-1)(0,0).
    const char* h1b = (const char*)h1;
    #pragma unroll 2
    for (int t = 0; t < 32; ++t) {
        int r  = wv * 8 + (t >> 2);
        int c0 = (t & 3) << 4;
        int c  = c0 + m;
        const char* base = h1b + ((r * IMG + c) * 32 + grp * 16 - 2080);
        bool pr = (r > 0), pl = (c > 0), pq = (c < 63);
        HU a;
        uint4 qa, qb;
        // j=0: taps (-1,-1), (-1,0)  (bias rides in as C)
        qa = ldt(base, 0,  pr && pl);
        qb = ldt(base, 32, pr);
        a.u[0] = qa.x; a.u[1] = qa.y; a.u[2] = qa.z; a.u[3] = qa.w;
        a.u[4] = qb.x; a.u[5] = qb.y; a.u[6] = qb.z; a.u[7] = qb.w;
        v8f acc = __builtin_amdgcn_wmma_f32_16x16x32_f16(false, a.h, false, bw2[0],
                                                         (short)0, cb2, false, false);
        // j=1: taps (-1,1), (0,-1)
        qa = ldt(base, 64,   pr && pq);
        qb = ldt(base, 2048, pl);
        a.u[0] = qa.x; a.u[1] = qa.y; a.u[2] = qa.z; a.u[3] = qa.w;
        a.u[4] = qb.x; a.u[5] = qb.y; a.u[6] = qb.z; a.u[7] = qb.w;
        acc = __builtin_amdgcn_wmma_f32_16x16x32_f16(false, a.h, false, bw2[1],
                                                     (short)0, acc, false, false);
        // j=2: tap (0,0) + zero pad
        qa = ldt(base, 2080, true);
        a.u[0] = qa.x; a.u[1] = qa.y; a.u[2] = qa.z; a.u[3] = qa.w;
        a.u[4] = 0u;   a.u[5] = 0u;   a.u[6] = 0u;   a.u[7] = 0u;
        acc = __builtin_amdgcn_wmma_f32_16x16x32_f16(false, a.h, false, bw2[2],
                                                     (short)0, acc, false, false);

        // conv2 relu + conv3 16-ch dot via DPP butterfly (all lanes, no LDS)
        float red[8];
        #pragma unroll
        for (int v = 0; v < 8; ++v) {
            float tv = relu1(acc[v]) * w3v;
            tv += dpp8f<SEL_XOR1>(tv);
            tv += dpp8f<SEL_XOR2>(tv);
            tv += dpp8f<SEL_XOR4>(tv);
            red[v] = ror8_addf(tv);            // cross 8-lane halves of the row
        }
        // single exec-masked region: sigmoid + two b128 stores (8 contiguous px)
        if (n == 0) {
            float* op = out + (size_t)b * (IMG * IMG) + r * IMG + c0 + grp * 8;
            float4 s0, s1;
            s0.x = 1.0f / (1.0f + __expf(-(red[0] + b3s)));
            s0.y = 1.0f / (1.0f + __expf(-(red[1] + b3s)));
            s0.z = 1.0f / (1.0f + __expf(-(red[2] + b3s)));
            s0.w = 1.0f / (1.0f + __expf(-(red[3] + b3s)));
            s1.x = 1.0f / (1.0f + __expf(-(red[4] + b3s)));
            s1.y = 1.0f / (1.0f + __expf(-(red[5] + b3s)));
            s1.z = 1.0f / (1.0f + __expf(-(red[6] + b3s)));
            s1.w = 1.0f / (1.0f + __expf(-(red[7] + b3s)));
            *(float4*)(op)     = s0;
            *(float4*)(op + 4) = s1;
        }
    }
}

extern "C" void kernel_launch(void* const* d_in, const int* in_sizes, int n_in,
                              void* d_out, int out_size, void* d_ws, size_t ws_size,
                              hipStream_t stream) {
    const float* x  = (const float*)d_in[0];
    const float* w1 = (const float*)d_in[1];
    const float* b1 = (const float*)d_in[2];
    const float* w2 = (const float*)d_in[3];
    const float* b2 = (const float*)d_in[4];
    const float* w3 = (const float*)d_in[5];
    const float* b3 = (const float*)d_in[6];
    float* out = (float*)d_out;
    size_t shmem = (size_t)XS2_BYTES + (size_t)H1_BYTES;   // ~151 KB
    pixelcnn_fused<<<1024, 256, shmem, stream>>>(x, w1, b1, w2, b2, w3, b3, out);
}
